// GraphTransformerLayer_74199855005743
// MI455X (gfx1250) — matrix operand order repaired and checked by reading, not compiled, not run on hardware
//
#include <hip/hip_runtime.h>
#include <hip/hip_bf16.h>
#include <math.h>

typedef __bf16 bf16;
typedef __attribute__((ext_vector_type(16))) __bf16 bf16x16;
typedef __attribute__((ext_vector_type(8)))  float  f32x8;
typedef unsigned int  u32;
typedef unsigned short u16;

#define NN  4096
#define DD  1024
#define HH  8
#define DHD 128

__device__ __forceinline__ f32x8 wmma_bf16(bf16x16 a, bf16x16 b, f32x8 c) {
  return __builtin_amdgcn_wmma_f32_16x16x32_bf16(false, a, false, b, (short)0, c,
                                                 false, false);
}

// ---- CDNA5 async global->LDS copy (16B per lane), tracked by ASYNCcnt ----
__device__ __forceinline__ void async_copy_b128(void* lds_dst, const void* gsrc) {
  u32 l = (u32)(size_t)lds_dst;  // low 32 bits of a shared pointer = LDS offset
  asm volatile("global_load_async_to_lds_b128 %0, %1, off"
               :: "v"(l), "v"(gsrc) : "memory");
}
__device__ __forceinline__ void wait_async0() {
  asm volatile("s_wait_asynccnt 0x0" ::: "memory");
}
__device__ __forceinline__ void wait_async4() {  // allow 4 newer ops in flight
  asm volatile("s_wait_asynccnt 0x4" ::: "memory");
}

// A fragment: 16x32 (MxK) tile. lane row = lane&15, kbase = (lane>>4)*8,
// elements at k = kbase+[0,8) and kbase+16+[0,8)  -> two b128 loads.
__device__ __forceinline__ bf16x16 load_a_frag(const bf16* __restrict__ p, int stride, int lane) {
  const int row = lane & 15;
  const int kb  = (lane >> 4) << 3;
  const bf16* r = p + row * stride + kb;
  union { bf16x16 v; uint4 q[2]; } f;
  f.q[0] = *(const uint4*)(r);
  f.q[1] = *(const uint4*)(r + 16);
  return f.v;
}

// B fragment from k-contiguous layout: B[k][n] = p[n*stride + k].
// lane col n = lane&15, kbase = (lane>>4)*8, k = kbase+[0,8) and kbase+16+[0,8) -> two b128 loads.
__device__ __forceinline__ bf16x16 load_bt_frag(const bf16* __restrict__ p, int stride, int lane) {
  const int n  = lane & 15;
  const int kb = (lane >> 4) << 3;
  const bf16* r = p + n * stride + kb;
  union { bf16x16 v; uint4 q[2]; } f;
  f.q[0] = *(const uint4*)(r);
  f.q[1] = *(const uint4*)(r + 16);
  return f.v;
}

// ------------------------------------------------------------- conversions
__global__ void k_cvt_bf16(const float* __restrict__ in, bf16* __restrict__ out, int n) {
  int i = blockIdx.x * blockDim.x + threadIdx.x;
  int stride = gridDim.x * blockDim.x;
  for (; i < n; i += stride) out[i] = (bf16)in[i];
}

// f32 (K x N, row-major) -> bf16 transposed (N x K): weight prep so GEMM B
// tiles are k-contiguous and can be staged with async DMA.
__global__ __launch_bounds__(256) void k_cvt_t(const float* __restrict__ in,
                                               bf16* __restrict__ out, int K, int N) {
  __shared__ float tile[32][33];
  const int tx = threadIdx.x & 31;
  const int ty = threadIdx.x >> 5;  // 0..7
  const int bk = blockIdx.y * 32;
  const int bn = blockIdx.x * 32;
#pragma unroll
  for (int i = 0; i < 4; ++i) {
    const int k = ty + i * 8;
    tile[k][tx] = in[(size_t)(bk + k) * N + bn + tx];
  }
  __syncthreads();
#pragma unroll
  for (int i = 0; i < 4; ++i) {
    const int n = ty + i * 8;
    out[(size_t)(bn + n) * K + bk + tx] = (bf16)tile[tx][n];
  }
}

// ---------------------------------------------------------------- GEMM
// C(MxNc) = A(MxK, bf16, row-major) @ B, with B given pre-transposed as
// Bt(Nc x K, bf16, k-contiguous). Double-buffered, fully-async staging.
#define BM 128
#define BN 128
#define BK 32
#define AS_STRIDE 40
#define BT_STRIDE 40

__device__ __forceinline__ float gelu_exact(float x) {
  return 0.5f * x * (1.0f + erff(x * 0.7071067811865475f));
}

template <int ACT, bool HAS_BIAS, bool HAS_RES, bool OUTF, bool OUTB, bool OUTB_T>
__global__ __launch_bounds__(256) void k_gemm_bf16(
    const bf16* __restrict__ A, const bf16* __restrict__ Bt,
    const float* __restrict__ bias, const float* __restrict__ residual,
    float* __restrict__ outF, bf16* __restrict__ outB,
    int M, int Nc, int K)
{
  __shared__ bf16 As[2][BM * AS_STRIDE];   // [m][k]
  __shared__ bf16 BsT[2][BN * BT_STRIDE];  // [n][k]

  const int t    = threadIdx.x;
  const int lane = t & 31;
  const int wave = t >> 5;
  const int wm   = (wave >> 2) * 64;  // 0 or 64
  const int wn   = (wave & 3) * 32;   // 0,32,64,96
  const int m0   = blockIdx.y * BM;
  const int n0   = blockIdx.x * BN;

  f32x8 acc[4][2] = {};

  const int s_row = t >> 1;        // 0..127: A row m / Bt row n
  const int s_ks  = (t & 1) * 16;

  auto stage = [&](int buf, int k0) {
    const bf16* ga = A + (size_t)(m0 + s_row) * K + k0 + s_ks;
    bf16* la = As[buf] + s_row * AS_STRIDE + s_ks;
    async_copy_b128(la, ga);
    async_copy_b128(la + 8, ga + 8);
    const bf16* gb = Bt + (size_t)(n0 + s_row) * K + k0 + s_ks;
    bf16* lb = BsT[buf] + s_row * BT_STRIDE + s_ks;
    async_copy_b128(lb, gb);
    async_copy_b128(lb + 8, gb + 8);
  };

  stage(0, 0);
  int cur = 0;
  for (int k0 = 0; k0 < K; k0 += BK) {
    if (k0 + BK < K) {          // prefetch next tile, then wait for current
      stage(cur ^ 1, k0 + BK);
      wait_async4();
    } else {
      wait_async0();
    }
    __syncthreads();

    bf16x16 af[4], bf_[2];
#pragma unroll
    for (int i = 0; i < 4; ++i)
      af[i] = load_a_frag(As[cur] + (wm + i * 16) * AS_STRIDE, AS_STRIDE, lane);
#pragma unroll
    for (int j = 0; j < 2; ++j)
      bf_[j] = load_bt_frag(BsT[cur] + (wn + j * 16) * BT_STRIDE, BT_STRIDE, lane);
#pragma unroll
    for (int i = 0; i < 4; ++i)
#pragma unroll
      for (int j = 0; j < 2; ++j)
        acc[i][j] = wmma_bf16(af[i], bf_[j], acc[i][j]);
    __syncthreads();
    cur ^= 1;
  }

  const int half = lane >> 4;
  const int nn   = lane & 15;
#pragma unroll
  for (int i = 0; i < 4; ++i) {
#pragma unroll
    for (int j = 0; j < 2; ++j) {
      const int col = n0 + wn + j * 16 + nn;
      const float bv = HAS_BIAS ? bias[col] : 0.0f;
#pragma unroll
      for (int r = 0; r < 8; ++r) {
        const int row = m0 + wm + i * 16 + r + 8 * half;
        float v = acc[i][j][r] + bv;
        if (ACT) v = gelu_exact(v);
        if (HAS_RES) v += residual[(size_t)row * Nc + col];
        if (OUTF) outF[(size_t)row * Nc + col] = v;
        if (OUTB) {
          if (OUTB_T) outB[(size_t)col * M + row] = (bf16)v;  // transposed out
          else        outB[(size_t)row * Nc + col] = (bf16)v;
        }
      }
    }
  }
}

// ---------------------------------------------------------------- attention
// One block (4 waves / 128 thr) = 64 query rows of one head, streaming softmax.
// K, V^T and adj tiles all staged by async DMA.
#define JB 64
#define KS_STRIDE 136   // Ks[key][dh]
#define VT_STRIDE 72    // Vt[dh][key]
#define PS_STRIDE 72    // Ps[row][key]
#define ADJ_STRIDE 68   // Adjs[qrow][key] f32

__global__ __launch_bounds__(128) void k_attn(
    const bf16* __restrict__ Qg, const bf16* __restrict__ Kg,
    const bf16* __restrict__ VgT,  // V^T: (D x N), dh-major rows
    const float* __restrict__ adj, const float* __restrict__ adj_bias,
    bf16* __restrict__ O)
{
  __shared__ bf16  Ks[JB * KS_STRIDE];     // 17408 B
  __shared__ bf16  Vt[DHD * VT_STRIDE];    // 18432 B
  __shared__ bf16  Ps[4][16 * PS_STRIDE];  //  9216 B
  __shared__ float Adjs[JB * ADJ_STRIDE];  // 17408 B   (total ~61 KB)

  const int t    = threadIdx.x;
  const int lane = t & 31;
  const int wave = t >> 5;
  const int h    = blockIdx.y;
  const int qb0  = blockIdx.x * 64;
  const int q0   = qb0 + wave * 16;  // 16 query rows per wave
  const int half = lane >> 4;
  const int nn   = lane & 15;
  const float beta  = adj_bias[h];
  const float scale = 0.08838834764831845f;  // 128^-0.5

  // persistent Q fragments (A-pattern directly from global, 2 x b128 each)
  bf16x16 qa[4];
#pragma unroll
  for (int ks = 0; ks < 4; ++ks)
    qa[ks] = load_a_frag(Qg + (size_t)q0 * DD + h * DHD + ks * 32, DD, lane);

  float mrow[8], lrow[8];
  f32x8 o_acc[8] = {};
#pragma unroll
  for (int r = 0; r < 8; ++r) { mrow[r] = -1e30f; lrow[r] = 0.0f; }

  for (int j0 = 0; j0 < NN; j0 += JB) {
    // K tile (64 x 128), V^T tile (128 x 64), adj tile (64 x 64 f32): all async
#pragma unroll
    for (int it = 0; it < 8; ++it) {
      const int e    = (it * 128 + t) * 8;
      const int krow = e >> 7;
      const int kcol = e & 127;
      async_copy_b128(Ks + krow * KS_STRIDE + kcol,
                      Kg + (size_t)(j0 + krow) * DD + h * DHD + kcol);
    }
#pragma unroll
    for (int it = 0; it < 8; ++it) {
      const int e    = (it * 128 + t) * 8;
      const int vrow = e >> 6;   // dh 0..127
      const int vcol = e & 63;   // key
      async_copy_b128(Vt + vrow * VT_STRIDE + vcol,
                      VgT + (size_t)(h * DHD + vrow) * NN + j0 + vcol);
    }
#pragma unroll
    for (int it = 0; it < 8; ++it) {
      const int e    = (it * 128 + t) * 4;
      const int arow = e >> 6;
      const int acol = e & 63;
      async_copy_b128(Adjs + arow * ADJ_STRIDE + acol,
                      adj + (size_t)(qb0 + arow) * NN + j0 + acol);
    }
    wait_async0();
    __syncthreads();

    // S = Q @ K^T (16x64); B[k=dh][n=key] = Ks[key][dh] -> k-contiguous
    f32x8 s[4] = {};
#pragma unroll
    for (int ks = 0; ks < 4; ++ks) {
#pragma unroll
      for (int nt = 0; nt < 4; ++nt) {
        bf16x16 bq = load_bt_frag(Ks + (nt * 16) * KS_STRIDE + ks * 32, KS_STRIDE, lane);
        s[nt] = wmma_bf16(qa[ks], bq, s[nt]);
      }
    }

    // online softmax per query row (row M = r + 8*half, cols in 16-lane group)
    const int lr0 = wave * 16;
#pragma unroll
    for (int r = 0; r < 8; ++r) {
      const int lr = lr0 + r + 8 * half;
      float lg[4];
      float mx = mrow[r];
#pragma unroll
      for (int nt = 0; nt < 4; ++nt) {
        const float a = Adjs[lr * ADJ_STRIDE + nt * 16 + nn];
        lg[nt] = s[nt][r] * scale + beta * a;
        mx = fmaxf(mx, lg[nt]);
      }
#pragma unroll
      for (int d = 1; d < 16; d <<= 1) mx = fmaxf(mx, __shfl_xor(mx, d, 32));
      const float sc = __expf(mrow[r] - mx);
      float ls = 0.0f;
      float pv[4];
#pragma unroll
      for (int nt = 0; nt < 4; ++nt) { pv[nt] = __expf(lg[nt] - mx); ls += pv[nt]; }
#pragma unroll
      for (int d = 1; d < 16; d <<= 1) ls += __shfl_xor(ls, d, 32);
      lrow[r] = lrow[r] * sc + ls;
      mrow[r] = mx;
#pragma unroll
      for (int n2 = 0; n2 < 8; ++n2) o_acc[n2][r] *= sc;
#pragma unroll
      for (int nt = 0; nt < 4; ++nt)
        Ps[wave][(r + 8 * half) * PS_STRIDE + nt * 16 + nn] = (bf16)pv[nt];
    }
    __syncthreads();

    // O += P @ V  (P as A-frag from per-wave slab; Vt gives k-contiguous B)
#pragma unroll
    for (int ks2 = 0; ks2 < 2; ++ks2) {
      bf16x16 pa = load_a_frag(Ps[wave] + ks2 * 32, PS_STRIDE, lane);
#pragma unroll
      for (int n2 = 0; n2 < 8; ++n2) {
        bf16x16 bv = load_bt_frag(Vt + (n2 * 16) * VT_STRIDE + ks2 * 32, VT_STRIDE, lane);
        o_acc[n2] = wmma_bf16(pa, bv, o_acc[n2]);
      }
    }
    __syncthreads();
  }

  // normalize & write bf16 attention output
#pragma unroll
  for (int r = 0; r < 8; ++r) {
    const float inv = 1.0f / lrow[r];
    const size_t row = (size_t)(q0 + r + 8 * half);
#pragma unroll
    for (int n2 = 0; n2 < 8; ++n2)
      O[row * DD + h * DHD + n2 * 16 + nn] = (bf16)(o_acc[n2][r] * inv);
  }
}

// ---------------------------------------------------------------- fused LN
__device__ __forceinline__ float block_sum(float v, float* red) {
#pragma unroll
  for (int d = 1; d < 32; d <<= 1) v += __shfl_xor(v, d, 32);
  const int wave = threadIdx.x >> 5;
  __syncthreads();
  if ((threadIdx.x & 31) == 0) red[wave] = v;
  __syncthreads();
  float s = 0.0f;
#pragma unroll
  for (int i = 0; i < 8; ++i) s += red[i];
  return s;
}

__global__ __launch_bounds__(256) void k_add_ln_ln(
    const float* __restrict__ x, const float* __restrict__ proj,
    const float* __restrict__ g1, const float* __restrict__ b1,
    const float* __restrict__ g2, const float* __restrict__ b2,
    float* __restrict__ h_out, bf16* __restrict__ f0_out)
{
  __shared__ float red[8];
  const int t = threadIdx.x;
  const size_t base = (size_t)blockIdx.x * DD;

  float v[4];
  float sum = 0.0f;
#pragma unroll
  for (int i = 0; i < 4; ++i) {
    const int c = t + i * 256;
    v[i] = x[base + c] + proj[base + c];
    sum += v[i];
  }
  sum = block_sum(sum, red);
  const float mu = sum * (1.0f / DD);
  float var = 0.0f;
#pragma unroll
  for (int i = 0; i < 4; ++i) { const float d = v[i] - mu; var += d * d; }
  var = block_sum(var, red) * (1.0f / DD);
  const float rstd = rsqrtf(var + 1e-5f);

  float hv[4];
  float sum2 = 0.0f;
#pragma unroll
  for (int i = 0; i < 4; ++i) {
    const int c = t + i * 256;
    hv[i] = (v[i] - mu) * rstd * g1[c] + b1[c];
    h_out[base + c] = hv[i];
    sum2 += hv[i];
  }
  sum2 = block_sum(sum2, red);
  const float mu2 = sum2 * (1.0f / DD);
  float var2 = 0.0f;
#pragma unroll
  for (int i = 0; i < 4; ++i) { const float d = hv[i] - mu2; var2 += d * d; }
  var2 = block_sum(var2, red) * (1.0f / DD);
  const float rstd2 = rsqrtf(var2 + 1e-5f);
#pragma unroll
  for (int i = 0; i < 4; ++i) {
    const int c = t + i * 256;
    f0_out[base + c] = (bf16)((hv[i] - mu2) * rstd2 * g2[c] + b2[c]);
  }
}

// ---------------------------------------------------------------- driver
static inline void launch_cvt(const float* in, bf16* out, int n, hipStream_t s) {
  int blocks = (n + 256 * 4 - 1) / (256 * 4);
  k_cvt_bf16<<<blocks, 256, 0, s>>>(in, out, n);
}
static inline void launch_cvt_t(const float* in, bf16* out, int K, int N, hipStream_t s) {
  k_cvt_t<<<dim3(N / 32, K / 32), 256, 0, s>>>(in, out, K, N);
}

extern "C" void kernel_launch(void* const* d_in, const int* in_sizes, int n_in,
                              void* d_out, int out_size, void* d_ws, size_t ws_size,
                              hipStream_t stream) {
  (void)in_sizes; (void)n_in; (void)out_size; (void)ws_size;
  const float* x     = (const float*)d_in[0];
  const float* adj   = (const float*)d_in[1];
  const float* Wq    = (const float*)d_in[2];
  const float* Wk    = (const float*)d_in[3];
  const float* Wv    = (const float*)d_in[4];
  const float* Wo_w  = (const float*)d_in[5];
  const float* Wo_b  = (const float*)d_in[6];
  const float* adj_b = (const float*)d_in[7];
  const float* ln1_g = (const float*)d_in[8];
  const float* ln1_b = (const float*)d_in[9];
  const float* ffn_g = (const float*)d_in[10];
  const float* ffn_b = (const float*)d_in[11];
  const float* w1    = (const float*)d_in[12];
  const float* b1    = (const float*)d_in[13];
  const float* w2    = (const float*)d_in[14];
  const float* b2    = (const float*)d_in[15];
  float* out = (float*)d_out;

  char* ws = (char*)d_ws;
  size_t off = 0;
  auto alloc = [&](size_t bytes) -> char* {
    char* p = ws + off;
    off += (bytes + 255) & ~(size_t)255;
    return p;
  };
  bf16* xb    = (bf16*)alloc((size_t)NN * DD * 2);
  bf16* wqb   = (bf16*)alloc((size_t)DD * DD * 2);       // transposed
  bf16* wkb   = (bf16*)alloc((size_t)DD * DD * 2);       // transposed
  bf16* wvb   = (bf16*)alloc((size_t)DD * DD * 2);       // transposed
  bf16* wob   = (bf16*)alloc((size_t)DD * DD * 2);       // transposed
  bf16* w1b   = (bf16*)alloc((size_t)DD * 2 * DD * 2);   // transposed (2D x D)
  bf16* w2b   = (bf16*)alloc((size_t)2 * DD * DD * 2);   // transposed (D x 2D)
  bf16* Qb    = (bf16*)alloc((size_t)NN * DD * 2);
  bf16* Kb    = (bf16*)alloc((size_t)NN * DD * 2);
  bf16* VbT   = (bf16*)alloc((size_t)NN * DD * 2);       // V^T (D x N)
  bf16* attb  = (bf16*)alloc((size_t)NN * DD * 2);
  float* projf = (float*)alloc((size_t)NN * DD * 4);
  float* hf    = (float*)alloc((size_t)NN * DD * 4);
  bf16* f0b   = (bf16*)alloc((size_t)NN * DD * 2);
  bf16* g1b   = (bf16*)alloc((size_t)NN * 2 * DD * 2);

  launch_cvt(x, xb, NN * DD, stream);
  launch_cvt_t(Wq,   wqb, DD, DD, stream);
  launch_cvt_t(Wk,   wkb, DD, DD, stream);
  launch_cvt_t(Wv,   wvb, DD, DD, stream);
  launch_cvt_t(Wo_w, wob, DD, DD, stream);
  launch_cvt_t(w1,   w1b, DD, 2 * DD, stream);
  launch_cvt_t(w2,   w2b, 2 * DD, DD, stream);

  dim3 gqkv(DD / BN, NN / BM);
  // Q,K: row-major bf16 out; V: transposed bf16 out
  k_gemm_bf16<0, false, false, false, true, false><<<gqkv, 256, 0, stream>>>(
      xb, wqb, nullptr, nullptr, nullptr, Qb, NN, DD, DD);
  k_gemm_bf16<0, false, false, false, true, false><<<gqkv, 256, 0, stream>>>(
      xb, wkb, nullptr, nullptr, nullptr, Kb, NN, DD, DD);
  k_gemm_bf16<0, false, false, false, true, true><<<gqkv, 256, 0, stream>>>(
      xb, wvb, nullptr, nullptr, nullptr, VbT, NN, DD, DD);

  // streaming-softmax attention
  k_attn<<<dim3(NN / 64, HH), 128, 0, stream>>>(Qb, Kb, VbT, adj, adj_b, attb);

  // output projection (+Wo_b) -> f32
  k_gemm_bf16<0, true, false, true, false, false><<<gqkv, 256, 0, stream>>>(
      attb, wob, Wo_b, nullptr, projf, nullptr, NN, DD, DD);

  // h = LN1(x + proj); f0 = LN_ffn(h) (bf16)
  k_add_ln_ln<<<NN, 256, 0, stream>>>(x, projf, ln1_g, ln1_b, ffn_g, ffn_b, hf, f0b);

  // FFN
  dim3 gf1(2 * DD / BN, NN / BM);
  k_gemm_bf16<1, true, false, false, true, false><<<gf1, 256, 0, stream>>>(
      f0b, w1b, b1, nullptr, nullptr, g1b, NN, 2 * DD, DD);
  dim3 gf2(DD / BN, NN / BM);
  k_gemm_bf16<0, true, true, true, false, false><<<gf2, 256, 0, stream>>>(
      g1b, w2b, b2, hf, out, nullptr, NN, DD, 2 * DD);
}
